// SocketGNN_4612794876407
// MI455X (gfx1250) — matrix operand order repaired and checked by reading, not compiled, not run on hardware
//
#include <hip/hip_runtime.h>

// ---------------- problem constants (from reference) ----------------
#define NSN   20000
#define NCN   10000
#define ESSN  320000
#define ECSN  160000
#define HIDD  128
#define NHEAD 4
#define EPG   8        // edges per 128-lane group in edge_aggr

typedef __attribute__((ext_vector_type(16))) __bf16        v16bf;
typedef __attribute__((ext_vector_type(8)))  float         v8f;
typedef __attribute__((ext_vector_type(8)))  unsigned int  v8u;

static __device__ __forceinline__ unsigned short f2bf(float f) {
  unsigned u = __float_as_uint(f);
  u += 0x7fffu + ((u >> 16) & 1u);           // round-to-nearest-even
  return (unsigned short)(u >> 16);
}
// monotonic float<->uint encoding so float max == uint max (for atomicMax)
static __device__ __forceinline__ unsigned fenc(float f) {
  unsigned u = __float_as_uint(f);
  return (u & 0x80000000u) ? ~u : (u | 0x80000000u);
}
static __device__ __forceinline__ float fdec(unsigned k) {
  unsigned u = (k & 0x80000000u) ? (k ^ 0x80000000u) : ~k;
  return __uint_as_float(u);
}

// ---------------- weight packing into WMMA B-fragment order ----------------
// Packed layout: elem[((tile*KC + kc)*32 + lane)*16 + t] = bf16(W[k, n])
//   n = tile*16 + (lane & 15),  k = kc*32 + (lane>>4)*16 + t
// (contiguous k order chosen here, same for A (not the ISA table order): lanes 0-15 hold K=0..15, lanes 16-31 K=16..31,
//  so each lane's 16 values are K-contiguous -> 2x b128 load per fragment)
__global__ void pack_w_kernel(const float* __restrict__ W,
                              unsigned short* __restrict__ out,
                              int K, int NCOL) {
  int idx = blockIdx.x * 256 + threadIdx.x;
  if (idx >= K * NCOL) return;
  int KC = K >> 5;
  int t    = idx & 15;
  int lane = (idx >> 4) & 31;
  int kc   = (idx >> 9) % KC;
  int tile = (idx >> 9) / KC;
  int n = tile * 16 + (lane & 15);
  int k = kc * 32 + (lane >> 4) * 16 + t;
  out[idx] = f2bf(W[(size_t)k * NCOL + n]);
}

// ---------------- WMMA bf16 GEMM: Y[N,NCOL] = act(X[N,K] @ W + bias) -------
// 256 threads = 8 wave32s; block owns 16 rows; each wave loops over 16-col tiles.
// Optional gather mode: row features = concat(featA[gsrc[row]], featB[gdst[row]])
// (used for the readout MLP so edge_feat [160k,256] is never materialized).
__global__ __launch_bounds__(256) void gemm_bf16_kernel(
    const float* __restrict__ X, const unsigned short* __restrict__ Wp,
    float* __restrict__ Y, int K, int NCOL,
    const float* __restrict__ bias, int dorelu,
    const int* __restrict__ gsrc, const int* __restrict__ gdst,
    const float* __restrict__ featA, const float* __restrict__ featB) {
  __shared__ alignas(16) unsigned short As[16 * 264];  // K<=256, +8 pad vs bank conflicts
  const int tid = threadIdx.x;
  const int rowbase = blockIdx.x * 16;
  const int Kpad = K + 8;

  // stage A tile fp32 -> bf16 into LDS
  for (int idx = tid; idx < 16 * K; idx += 256) {
    int r = idx / K, k = idx - r * K;
    int grow = rowbase + r;
    float v;
    if (gsrc) {
      v = (k < HIDD) ? featA[(size_t)gsrc[grow] * HIDD + k]
                     : featB[(size_t)gdst[grow] * HIDD + (k - HIDD)];
    } else {
      v = X[(size_t)grow * K + k];
    }
    As[r * Kpad + k] = f2bf(v);
  }
  __syncthreads();

  const int wave = tid >> 5, lane = tid & 31;
  const int half = lane >> 4, nlan = lane & 15;
  const int KC = K >> 5, NT = NCOL >> 4;
  const unsigned short* arow = As + nlan * Kpad;

  for (int tile = wave; tile < NT; tile += 8) {
    float bv = bias ? bias[tile * 16 + nlan] : 0.0f;
    v8f acc;
#pragma unroll
    for (int i = 0; i < 8; ++i) acc[i] = bv;
    const unsigned short* wp =
        Wp + ((size_t)tile * KC) * 32 * 16 + (size_t)lane * 16;
    for (int kc = 0; kc < KC; ++kc) {
      // A fragment: lane holds row (lane&15); K = half*16+[0,16)
      const uint4* ap = (const uint4*)(arow + kc * 32 + half * 16);
      uint4 a0 = ap[0], a1 = ap[1];
      // B fragment: 32 contiguous bytes per lane from packed weights (L2-hot)
      const uint4* bp = (const uint4*)(wp + (size_t)kc * 32 * 16);
      uint4 b0 = bp[0], b1 = bp[1];
      v8u au, bu;
      au[0] = a0.x; au[1] = a0.y; au[2] = a0.z; au[3] = a0.w;
      au[4] = a1.x; au[5] = a1.y; au[6] = a1.z; au[7] = a1.w;
      bu[0] = b0.x; bu[1] = b0.y; bu[2] = b0.z; bu[3] = b0.w;
      bu[4] = b1.x; bu[5] = b1.y; bu[6] = b1.z; bu[7] = b1.w;
      v16bf a = __builtin_bit_cast(v16bf, au);
      v16bf b = __builtin_bit_cast(v16bf, bu);
      acc = __builtin_amdgcn_wmma_f32_16x16x32_bf16(
          false, a, false, b, (short)0, acc, false, false);
    }
    // C/D layout: lane col = lane&15, rows = half*8 + r
    int n = tile * 16 + nlan;
#pragma unroll
    for (int r = 0; r < 8; ++r) {
      float v = acc[r];
      if (dorelu) v = fmaxf(v, 0.0f);
      Y[((size_t)(rowbase + half * 8 + r)) * NCOL + n] = v;
    }
  }
}

// ---------------- small dense kernels ----------------
__global__ void proj_kernel(const float* __restrict__ x, const float* __restrict__ W,
                            const float* __restrict__ b, float* __restrict__ y,
                            int N, int K) {
  int idx = blockIdx.x * 256 + threadIdx.x;
  if (idx >= N * HIDD) return;
  int n = idx >> 7, o = idx & 127;
  float a = b[o];
  for (int k = 0; k < K; ++k) a += x[(size_t)n * K + k] * W[(size_t)k * HIDD + o];
  y[idx] = a;
}

// wv[k*4+h] = sum_c W[k, h*128+c] * att[h, c]   (folds hd GEMM into a GEMV)
__global__ void wv_kernel(const float* __restrict__ W, const float* __restrict__ att,
                          float* __restrict__ wv) {
  int idx = blockIdx.x * 256 + threadIdx.x;
  if (idx >= HIDD * NHEAD) return;
  int k = idx >> 2, h = idx & 3;
  float a = 0.f;
  for (int c = 0; c < HIDD; ++c)
    a += W[(size_t)k * (NHEAD * HIDD) + h * HIDD + c] * att[h * HIDD + c];
  wv[idx] = a;
}

__global__ void score_kernel(const float* __restrict__ x, const float* __restrict__ wv,
                             float* __restrict__ out, int N) {
  int idx = blockIdx.x * 256 + threadIdx.x;
  if (idx >= N * NHEAD) return;
  int n = idx >> 2, h = idx & 3;
  float a = 0.f;
  for (int k = 0; k < HIDD; ++k) a += x[(size_t)n * HIDD + k] * wv[k * 4 + h];
  out[idx] = a;
}

__global__ void fill_enc_kernel(unsigned* p, float v, int n) {
  int i = blockIdx.x * 256 + threadIdx.x;
  if (i < n) p[i] = fenc(v);
}
__global__ void fill_f32_kernel(float* p, float v, int n) {
  int i = blockIdx.x * 256 + threadIdx.x;
  if (i < n) p[i] = v;
}
__global__ void bias_init_kernel(float* __restrict__ acc, const float* __restrict__ b1,
                                 const float* __restrict__ b2, int N) {
  int idx = blockIdx.x * 256 + threadIdx.x;
  if (idx >= N * HIDD) return;
  int c = idx & 127;
  acc[idx] = b1[c] + (b2 ? b2[c] : 0.0f);
}
__global__ void relu_kernel(const float* __restrict__ x, float* __restrict__ y, int n) {
  int i = blockIdx.x * 256 + threadIdx.x;
  if (i < n) y[i] = fmaxf(x[i], 0.0f);
}

// ---------------- edge kernels (segment softmax + aggregation) ----------------
__global__ void edge_max_kernel(const float* __restrict__ a_src,
                                const float* __restrict__ a_dst,
                                const int* __restrict__ src, const int* __restrict__ dst,
                                int E, unsigned* __restrict__ menc) {
  int e = blockIdx.x * 256 + threadIdx.x;
  if (e >= E) return;
  int si = src[e], di = dst[e];
#pragma unroll
  for (int h = 0; h < NHEAD; ++h) {
    float v = a_src[si * 4 + h] + a_dst[di * 4 + h];
    v = v > 0.f ? v : 0.2f * v;                  // leaky_relu(0.2)
    atomicMax(&menc[di * 4 + h], fenc(v));       // global_atomic_max_u32
  }
}

__global__ void edge_exp_kernel(const float* __restrict__ a_src,
                                const float* __restrict__ a_dst,
                                const int* __restrict__ src, const int* __restrict__ dst,
                                int E, const unsigned* __restrict__ menc,
                                float* __restrict__ exb, float* __restrict__ den) {
  int e = blockIdx.x * 256 + threadIdx.x;
  if (e >= E) return;
  int si = src[e], di = dst[e];
#pragma unroll
  for (int h = 0; h < NHEAD; ++h) {
    float v = a_src[si * 4 + h] + a_dst[di * 4 + h];
    v = v > 0.f ? v : 0.2f * v;
    float ex = __expf(v - fdec(menc[di * 4 + h]));
    exb[(size_t)e * 4 + h] = ex;
    atomicAdd(&den[di * 4 + h], ex);
  }
}

// Each 128-lane group walks EPG edges, prefetching the next edge's hs row
// (global_prefetch_b8) so the random 2KB row fetch overlaps the current
// edge's compute + atomics. This phase is the HBM/L2 roofline limiter.
__global__ void edge_aggr_kernel(const float* __restrict__ exb,
                                 const float* __restrict__ den,
                                 const float* __restrict__ hs,
                                 const int* __restrict__ src, const int* __restrict__ dst,
                                 int E, float* __restrict__ acc) {
  int group = blockIdx.x * 2 + (threadIdx.x >> 7);
  int c = threadIdx.x & 127;
  int e0 = group * EPG;
#pragma unroll 1
  for (int i = 0; i < EPG; ++i) {
    int e = e0 + i;
    if (e >= E) return;
    int si = src[e], di = dst[e];
    if (i + 1 < EPG && e + 1 < E) {
      size_t nrow = (size_t)src[e + 1] * (NHEAD * HIDD);
#pragma unroll
      for (int h = 0; h < NHEAD; ++h)
        __builtin_prefetch(&hs[nrow + h * HIDD + c], 0, 3);
    }
    float a = 0.f;
#pragma unroll
    for (int h = 0; h < NHEAD; ++h) {
      float alpha = exb[(size_t)e * 4 + h] / (den[di * 4 + h] + 1e-16f);
      a += alpha * hs[(size_t)si * (NHEAD * HIDD) + h * HIDD + c];
    }
    atomicAdd(&acc[(size_t)di * HIDD + c], 0.25f * a);  // head mean folded in
  }
}

// final tiny layer: out[e, j] = b3[j] + h2[e,:] @ w3[:, j]
__global__ void head_kernel(const float* __restrict__ h2, const float* __restrict__ w3,
                            const float* __restrict__ b3, float* __restrict__ out,
                            int E, int OUT) {
  int e = blockIdx.x * 256 + threadIdx.x;
  if (e >= E) return;
  for (int j = 0; j < OUT; ++j) {
    float a = b3[j];
    for (int k = 0; k < 64; ++k) a += h2[(size_t)e * 64 + k] * w3[k * OUT + j];
    out[(size_t)e * OUT + j] = a;
  }
}

// ---------------- host orchestration ----------------
extern "C" void kernel_launch(void* const* d_in, const int* in_sizes, int n_in,
                              void* d_out, int out_size, void* d_ws, size_t ws_size,
                              hipStream_t stream) {
  (void)in_sizes; (void)n_in; (void)out_size; (void)ws_size;
  auto cdiv = [](int a, int b) { return (a + b - 1) / b; };
  auto F = [&](int i) { return (const float*)d_in[i]; };

  const float* surface_x = F(0);
  const float* component_x = F(1);
  // params pytree flattened in JAX sorted-key order:
  // cls(b1,b2,b3,w1,w2,w3), comp_proj_b, comp_proj_w,
  // layers[0..2]{cs,sc,ss}{att_dst,att_src,bias,w_dst,w_src},
  // reg(b1,b2,b3,w1,w2,w3), surf_proj_b, surf_proj_w
  const int P = 2;
  const float *cls_b1 = F(P + 0), *cls_b2 = F(P + 1), *cls_b3 = F(P + 2);
  const float *cls_w1 = F(P + 3), *cls_w2 = F(P + 4), *cls_w3 = F(P + 5);
  const float *comp_b = F(P + 6), *comp_w = F(P + 7);
  const float *reg_b1 = F(P + 53), *reg_b2 = F(P + 54), *reg_b3 = F(P + 55);
  const float *reg_w1 = F(P + 56), *reg_w2 = F(P + 57), *reg_w3 = F(P + 58);
  const float *surf_b = F(P + 59), *surf_w = F(P + 60);
  const int* edge_ss  = (const int*)d_in[P + 61];  // [2, ESS]
  const int* e_cs_src = (const int*)d_in[P + 62];
  const int* e_cs_dst = (const int*)d_in[P + 63];
  const int* e_sc_src = (const int*)d_in[P + 64];
  const int* e_sc_dst = (const int*)d_in[P + 65];

  // workspace carve-out
  char* wsb = (char*)d_ws;
  size_t off = 0;
  auto carve = [&](size_t bytes) {
    void* p = wsb + off;
    off += (bytes + 255) & ~(size_t)255;
    return p;
  };
  float* sfeat = (float*)carve((size_t)NSN * HIDD * 4);
  float* cfeat = (float*)carve((size_t)NCN * HIDD * 4);
  float* s_acc = (float*)carve((size_t)NSN * HIDD * 4);
  float* c_acc = (float*)carve((size_t)NCN * HIDD * 4);
  float* big   = (float*)carve((size_t)ECSN * HIDD * 4);  // hs [N,512] / h1 [E,128]
  float* h2    = (float*)carve((size_t)ECSN * 64 * 4);
  float* a_src = (float*)carve((size_t)NSN * 4 * 4);
  float* a_dst = (float*)carve((size_t)NSN * 4 * 4);
  unsigned* menc = (unsigned*)carve((size_t)NSN * 4 * 4);
  float* den   = (float*)carve((size_t)NSN * 4 * 4);
  float* exb   = (float*)carve((size_t)ESSN * 4 * 4);
  unsigned short* wpack = (unsigned short*)carve((size_t)HIDD * 512 * 2);
  float* wv    = (float*)carve((size_t)HIDD * 4 * 4 * 2);

  auto run_gat = [&](const float* xs, int Nsrc, const float* xd, int Ndst,
                     const int* src, const int* dst, int E,
                     const float* w_src, const float* w_dst,
                     const float* att_src, const float* att_dst, float* acc) {
    pack_w_kernel<<<cdiv(HIDD * 512, 256), 256, 0, stream>>>(w_src, wpack, HIDD, 512);
    gemm_bf16_kernel<<<Nsrc / 16, 256, 0, stream>>>(
        xs, wpack, big, HIDD, 512, nullptr, 0, nullptr, nullptr, nullptr, nullptr);
    wv_kernel<<<cdiv(HIDD * 4, 256), 256, 0, stream>>>(w_src, att_src, wv);
    wv_kernel<<<cdiv(HIDD * 4, 256), 256, 0, stream>>>(w_dst, att_dst, wv + 512);
    score_kernel<<<cdiv(Nsrc * 4, 256), 256, 0, stream>>>(xs, wv, a_src, Nsrc);
    score_kernel<<<cdiv(Ndst * 4, 256), 256, 0, stream>>>(xd, wv + 512, a_dst, Ndst);
    fill_enc_kernel<<<cdiv(Ndst * 4, 256), 256, 0, stream>>>(menc, -3.0e38f, Ndst * 4);
    fill_f32_kernel<<<cdiv(Ndst * 4, 256), 256, 0, stream>>>(den, 0.f, Ndst * 4);
    edge_max_kernel<<<cdiv(E, 256), 256, 0, stream>>>(a_src, a_dst, src, dst, E, menc);
    edge_exp_kernel<<<cdiv(E, 256), 256, 0, stream>>>(a_src, a_dst, src, dst, E, menc, exb, den);
    edge_aggr_kernel<<<cdiv(E, 2 * EPG), 256, 0, stream>>>(exb, den, big, src, dst, E, acc);
  };

  // input projections
  proj_kernel<<<cdiv(NSN * HIDD, 256), 256, 0, stream>>>(surface_x, surf_w, surf_b, sfeat, NSN, 13);
  proj_kernel<<<cdiv(NCN * HIDD, 256), 256, 0, stream>>>(component_x, comp_w, comp_b, cfeat, NCN, 56);

  for (int l = 0; l < 3; ++l) {
    int L = P + 8 + l * 15;
    const float *cs_attd = F(L + 0), *cs_atts = F(L + 1), *cs_bias = F(L + 2),
                *cs_wd = F(L + 3), *cs_ws = F(L + 4);
    const float *sc_attd = F(L + 5), *sc_atts = F(L + 6), *sc_bias = F(L + 7),
                *sc_wd = F(L + 8), *sc_ws = F(L + 9);
    const float *ss_attd = F(L + 10), *ss_atts = F(L + 11), *ss_bias = F(L + 12),
                *ss_wd = F(L + 13), *ss_ws = F(L + 14);

    bias_init_kernel<<<cdiv(NSN * HIDD, 256), 256, 0, stream>>>(s_acc, ss_bias, cs_bias, NSN);
    bias_init_kernel<<<cdiv(NCN * HIDD, 256), 256, 0, stream>>>(c_acc, sc_bias, nullptr, NCN);

    run_gat(sfeat, NSN, sfeat, NSN, edge_ss, edge_ss + ESSN, ESSN,
            ss_ws, ss_wd, ss_atts, ss_attd, s_acc);
    run_gat(cfeat, NCN, sfeat, NSN, e_cs_src, e_cs_dst, ECSN,
            cs_ws, cs_wd, cs_atts, cs_attd, s_acc);
    run_gat(sfeat, NSN, cfeat, NCN, e_sc_src, e_sc_dst, ECSN,
            sc_ws, sc_wd, sc_atts, sc_attd, c_acc);

    relu_kernel<<<cdiv(NSN * HIDD, 256), 256, 0, stream>>>(s_acc, sfeat, NSN * HIDD);
    relu_kernel<<<cdiv(NCN * HIDD, 256), 256, 0, stream>>>(c_acc, cfeat, NCN * HIDD);
  }

  // readout MLPs over edge features (gathered on the fly, never materialized)
  float* outp = (float*)d_out;
  // classifier head -> d_out[0 .. ECS)
  pack_w_kernel<<<cdiv(256 * 128, 256), 256, 0, stream>>>(cls_w1, wpack, 256, 128);
  gemm_bf16_kernel<<<ECSN / 16, 256, 0, stream>>>(
      nullptr, wpack, big, 256, 128, cls_b1, 1, e_cs_src, e_cs_dst, cfeat, sfeat);
  pack_w_kernel<<<cdiv(128 * 64, 256), 256, 0, stream>>>(cls_w2, wpack, 128, 64);
  gemm_bf16_kernel<<<ECSN / 16, 256, 0, stream>>>(
      big, wpack, h2, 128, 64, cls_b2, 1, nullptr, nullptr, nullptr, nullptr);
  head_kernel<<<cdiv(ECSN, 256), 256, 0, stream>>>(h2, cls_w3, cls_b3, outp, ECSN, 1);
  // regression head -> d_out[ECS .. ECS+3*ECS)
  pack_w_kernel<<<cdiv(256 * 128, 256), 256, 0, stream>>>(reg_w1, wpack, 256, 128);
  gemm_bf16_kernel<<<ECSN / 16, 256, 0, stream>>>(
      nullptr, wpack, big, 256, 128, reg_b1, 1, e_cs_src, e_cs_dst, cfeat, sfeat);
  pack_w_kernel<<<cdiv(128 * 64, 256), 256, 0, stream>>>(reg_w2, wpack, 128, 64);
  gemm_bf16_kernel<<<ECSN / 16, 256, 0, stream>>>(
      big, wpack, h2, 128, 64, reg_b2, 1, nullptr, nullptr, nullptr, nullptr);
  head_kernel<<<cdiv(ECSN, 256), 256, 0, stream>>>(h2, reg_w3, reg_b3, outp + ECSN, ECSN, 3);
}
